// WindowAttention3D_17386027614782
// MI455X (gfx1250) — matrix operand order repaired and checked by reading, not compiled, not run on hardware
//
#include <hip/hip_runtime.h>

#define N_TOK 343
#define NPAD  352
#define DIMC  96
#define HD    32
#define NHEADS 3
#define VTS   360   // V^T row stride (352 + 8 pad)
#define SCALE 0.17677669529663687f

typedef __attribute__((ext_vector_type(16))) __bf16 v16bf;
typedef __attribute__((ext_vector_type(8)))  __bf16 v8bf;
typedef __attribute__((ext_vector_type(8)))  float  v8f;

// ---- WMMA bf16 16x16x32 fragment loaders (wave32) ----
// A (16x32), 16-bit layout: lane m=lane&15 -> row m; hi=lane>>4:
//   elems 0..7  hold K = k0 + 8*hi + 0..7
//   elems 8..15 hold K = k0 + 16 + 8*hi + 0..7
__device__ __forceinline__ v16bf a_frag_lds(const __bf16* base, int stride,
                                            int row0, int k0, int lane) {
  const int m = lane & 15, hi = lane >> 4;
  const __bf16* p = base + (row0 + m) * stride + k0 + 8 * hi;
  const v8bf c0 = *(const v8bf*)(p);
  const v8bf c1 = *(const v8bf*)(p + 16);
  v16bf r;
#pragma unroll
  for (int i = 0; i < 8; ++i) { r[i] = c0[i]; r[8 + i] = c1[i]; }
  return r;
}

// B (32x16) loaded from row-major B^T (16 rows x K):
//   lane n=lane&15 -> column n; elems 0..7 hold K = k0 + 8*hi + 0..7, elems 8..15 hold K = k0 + 16 + 8*hi + 0..7
__device__ __forceinline__ v16bf b_frag_lds(const __bf16* base, int stride,
                                            int row0, int k0, int lane) {
  const int n = lane & 15, hi = lane >> 4;
  const __bf16* p = base + (row0 + n) * stride + k0 + 8 * hi;
  const v8bf c0 = *(const v8bf*)(p);
  const v8bf c1 = *(const v8bf*)(p + 16);
  v16bf r;
#pragma unroll
  for (int i = 0; i < 8; ++i) { r[i] = c0[i]; r[8 + i] = c1[i]; }
  return r;
}

// A fragment from global f32 (row pointer precomputed), cvt -> bf16
__device__ __forceinline__ v16bf a_frag_f32g(const float* rowp, int k0,
                                             int lane, bool valid) {
  v16bf r;
  const int hi = lane >> 4;
  if (valid) {
    const float* p = rowp + k0 + 8 * hi;
    const float4 a0 = *(const float4*)(p);
    const float4 a1 = *(const float4*)(p + 4);
    const float4 b0 = *(const float4*)(p + 16);
    const float4 b1 = *(const float4*)(p + 20);
    r[0]=(__bf16)a0.x; r[1]=(__bf16)a0.y; r[2]=(__bf16)a0.z; r[3]=(__bf16)a0.w;
    r[4]=(__bf16)a1.x; r[5]=(__bf16)a1.y; r[6]=(__bf16)a1.z; r[7]=(__bf16)a1.w;
    r[8]=(__bf16)b0.x; r[9]=(__bf16)b0.y; r[10]=(__bf16)b0.z; r[11]=(__bf16)b0.w;
    r[12]=(__bf16)b1.x; r[13]=(__bf16)b1.y; r[14]=(__bf16)b1.z; r[15]=(__bf16)b1.w;
  } else {
#pragma unroll
    for (int i = 0; i < 16; ++i) r[i] = (__bf16)0.0f;
  }
  return r;
}

// A fragment from global bf16 (row pointer precomputed)
__device__ __forceinline__ v16bf a_frag_b16g(const __bf16* rowp, int k0,
                                             int lane, bool valid) {
  v16bf r;
  const int hi = lane >> 4;
  if (valid) {
    const __bf16* p = rowp + k0 + 8 * hi;
    const v8bf c0 = *(const v8bf*)(p);
    const v8bf c1 = *(const v8bf*)(p + 16);
#pragma unroll
    for (int i = 0; i < 8; ++i) { r[i] = c0[i]; r[8 + i] = c1[i]; }
  } else {
#pragma unroll
    for (int i = 0; i < 16; ++i) r[i] = (__bf16)0.0f;
  }
  return r;
}

// ======================= Kernel A: QKV + flash attention ====================
__global__ __launch_bounds__(128, 1) void wa3d_attn_kernel(
    const float* __restrict__ x, const float* __restrict__ mask,
    const float* __restrict__ qkv_w, const float* __restrict__ qkv_b,
    const float* __restrict__ bias_table, const int* __restrict__ rel_index,
    __bf16* __restrict__ cxt) {
  extern __shared__ __align__(16) __bf16 smem[];
  __bf16* wlds = smem;                 // [96][96]  per-head qkv weight rows
  __bf16* qs   = wlds + 96 * DIMC;     // [352][32] Q * scale
  __bf16* ks   = qs + NPAD * HD;       // [352][32] K
  __bf16* vt   = ks + NPAD * HD;       // [32][360] V^T
  __bf16* pbuf = vt + HD * VTS;        // [4][16][32] per-wave P bounce

  const int blk = blockIdx.x;
  const int b = blk / NHEADS, h = blk % NHEADS;
  const int tid = threadIdx.x, lane = tid & 31, wv = tid >> 5;
  const int m16 = lane & 15, hi = lane >> 4;

  // Stage qkv weight rows for this head (sel=0:q,1:k,2:v) -> wlds[sel*32+r][c]
  for (int idx = tid; idx < 96 * DIMC; idx += 128) {
    const int r = idx / DIMC, c = idx - r * DIMC;
    const int sel = r >> 5, rl = r & 31;
    wlds[idx] = (__bf16)qkv_w[(sel * DIMC + h * HD + rl) * DIMC + c];
  }
  __syncthreads();

  // ---- QKV GEMM: (352x96) x (96x32) per sel via WMMA bf16 ----
  const float* xw = x + (size_t)b * N_TOK * DIMC;
  for (int t = wv; t < 3 * 22 * 2; t += 4) {
    const int sel = t / 44, rem = t - sel * 44, mt = rem >> 1, nt = rem & 1;
    const int row = mt * 16 + m16;
    v8f acc = {};
#pragma unroll
    for (int kk = 0; kk < 3; ++kk) {
      const v16bf a  = a_frag_f32g(xw + (size_t)row * DIMC, kk * 32, lane, row < N_TOK);
      const v16bf bm = b_frag_lds(wlds, DIMC, sel * 32 + nt * 16, kk * 32, lane);
      acc = __builtin_amdgcn_wmma_f32_16x16x32_bf16(false, a, false, bm,
                                                    (short)0, acc, false, false);
    }
    const int colb = nt * 16 + m16;
    const float bias = qkv_b[sel * DIMC + h * HD + colb];
#pragma unroll
    for (int i = 0; i < 8; ++i) {
      const int orow = mt * 16 + i + 8 * hi;   // D layout: VGPR i, lane halves
      const float v = acc[i] + bias;
      if (sel == 0)      qs[orow * HD + colb]  = (__bf16)(v * SCALE);
      else if (sel == 1) ks[orow * HD + colb]  = (__bf16)v;
      else               vt[colb * VTS + orow] = (__bf16)v;
    }
  }
  __syncthreads();

  // ---- Flash attention: waves own 16-row Q tiles, stream 32-col supertiles --
  const float* mw = mask + (size_t)b * N_TOK * N_TOK;
  __bf16* pw = pbuf + wv * (16 * 32);
  for (int rt = wv; rt < 22; rt += 4) {
    const v16bf aq = a_frag_lds(qs, HD, rt * 16, 0, lane);
    float mrow[8], lrow[8];
    v8f acc0 = {}, acc1 = {};
#pragma unroll
    for (int i = 0; i < 8; ++i) { mrow[i] = -1e30f; lrow[i] = 0.0f; }

    for (int ct = 0; ct < 11; ++ct) {
      const int c0 = ct * 32;
      const v16bf bk0 = b_frag_lds(ks, HD, c0, 0, lane);
      const v16bf bk1 = b_frag_lds(ks, HD, c0 + 16, 0, lane);
      const v8f z = {};
      v8f s0 = __builtin_amdgcn_wmma_f32_16x16x32_bf16(false, aq, false, bk0,
                                                       (short)0, z, false, false);
      v8f s1 = __builtin_amdgcn_wmma_f32_16x16x32_bf16(false, aq, false, bk1,
                                                       (short)0, z, false, false);
      const int col0 = c0 + m16, col1 = col0 + 16;
#pragma unroll
      for (int i = 0; i < 8; ++i) {
        const int qi = rt * 16 + i + 8 * hi;
        const bool qv = qi < N_TOK;
        if (qv && (col0 < N_TOK))
          s0[i] += mw[qi * N_TOK + col0] +
                   bias_table[rel_index[qi * N_TOK + col0] * NHEADS + h];
        else s0[i] = -1e30f;
        if (qv && (col1 < N_TOK))
          s1[i] += mw[qi * N_TOK + col1] +
                   bias_table[rel_index[qi * N_TOK + col1] * NHEADS + h];
        else s1[i] = -1e30f;
      }
      // Online softmax; rows live across 16-lane half-groups in the C layout.
#pragma unroll
      for (int i = 0; i < 8; ++i) {
        float tmax = fmaxf(s0[i], s1[i]);
        tmax = fmaxf(tmax, __shfl_xor(tmax, 1, 32));
        tmax = fmaxf(tmax, __shfl_xor(tmax, 2, 32));
        tmax = fmaxf(tmax, __shfl_xor(tmax, 4, 32));
        tmax = fmaxf(tmax, __shfl_xor(tmax, 8, 32));
        const float mn = fmaxf(mrow[i], tmax);
        const float alpha = __expf(mrow[i] - mn);
        const float p0 = __expf(s0[i] - mn);
        const float p1 = __expf(s1[i] - mn);
        float rs = p0 + p1;
        rs += __shfl_xor(rs, 1, 32);
        rs += __shfl_xor(rs, 2, 32);
        rs += __shfl_xor(rs, 4, 32);
        rs += __shfl_xor(rs, 8, 32);
        lrow[i] = lrow[i] * alpha + rs;
        mrow[i] = mn;
        acc0[i] *= alpha;
        acc1[i] *= alpha;
        const int pr = i + 8 * hi;
        pw[pr * 32 + m16]      = (__bf16)p0;
        pw[pr * 32 + 16 + m16] = (__bf16)p1;
      }
      asm volatile("s_wait_dscnt 0" ::: "memory");  // C-layout -> A-layout bounce
      const v16bf ap  = a_frag_lds(pw, 32, 0, 0, lane);
      const v16bf bv0 = b_frag_lds(vt, VTS, 0, c0, lane);
      const v16bf bv1 = b_frag_lds(vt, VTS, 16, c0, lane);
      acc0 = __builtin_amdgcn_wmma_f32_16x16x32_bf16(false, ap, false, bv0,
                                                     (short)0, acc0, false, false);
      acc1 = __builtin_amdgcn_wmma_f32_16x16x32_bf16(false, ap, false, bv1,
                                                     (short)0, acc1, false, false);
    }

    __bf16* cw = cxt + (size_t)b * N_TOK * DIMC + h * HD;
#pragma unroll
    for (int i = 0; i < 8; ++i) {
      const int qi = rt * 16 + i + 8 * hi;
      if (qi < N_TOK) {
        const float inv = 1.0f / lrow[i];
        cw[qi * DIMC + m16]      = (__bf16)(acc0[i] * inv);
        cw[qi * DIMC + 16 + m16] = (__bf16)(acc1[i] * inv);
      }
    }
  }
}

// ======================= Kernel B: output projection ========================
__global__ __launch_bounds__(128, 1) void wa3d_proj_kernel(
    const __bf16* __restrict__ cxt, const float* __restrict__ proj_w,
    const float* __restrict__ proj_b, float* __restrict__ out) {
  __shared__ __align__(16) __bf16 pwl[DIMC * DIMC];
  const int b = blockIdx.x;
  const int tid = threadIdx.x, lane = tid & 31, wv = tid >> 5;
  const int m16 = lane & 15, hi = lane >> 4;

  for (int idx = tid; idx < DIMC * DIMC; idx += 128)
    pwl[idx] = (__bf16)proj_w[idx];
  __syncthreads();

  const __bf16* cw = cxt + (size_t)b * N_TOK * DIMC;
  float* ow = out + (size_t)b * N_TOK * DIMC;
  for (int t = wv; t < 22 * 6; t += 4) {
    const int mt = t / 6, nt = t - mt * 6;
    const int row = mt * 16 + m16;
    v8f acc = {};
#pragma unroll
    for (int kk = 0; kk < 3; ++kk) {
      const v16bf a  = a_frag_b16g(cw + (size_t)row * DIMC, kk * 32, lane, row < N_TOK);
      const v16bf bm = b_frag_lds(pwl, DIMC, nt * 16, kk * 32, lane);
      acc = __builtin_amdgcn_wmma_f32_16x16x32_bf16(false, a, false, bm,
                                                    (short)0, acc, false, false);
    }
    const int colb = nt * 16 + m16;
    const float pb = proj_b[colb];
#pragma unroll
    for (int i = 0; i < 8; ++i) {
      const int qi = mt * 16 + i + 8 * hi;
      if (qi < N_TOK) ow[qi * DIMC + colb] = acc[i] + pb;
    }
  }
}

// =============================== launcher ===================================
extern "C" void kernel_launch(void* const* d_in, const int* in_sizes, int n_in,
                              void* d_out, int out_size, void* d_ws, size_t ws_size,
                              hipStream_t stream) {
  const float* x          = (const float*)d_in[0];
  const float* mask       = (const float*)d_in[1];
  const float* qkv_w      = (const float*)d_in[2];
  const float* qkv_b      = (const float*)d_in[3];
  const float* proj_w     = (const float*)d_in[4];
  const float* proj_b     = (const float*)d_in[5];
  const float* bias_table = (const float*)d_in[6];
  const int*   rel_index  = (const int*)d_in[7];

  // workspace: per-head attention context, bf16, (512, 343, 96) = ~33.7 MB
  __bf16* cxt = (__bf16*)d_ws;

  const size_t smemA =
      (size_t)(96 * DIMC + 2 * NPAD * HD + HD * VTS + 4 * 16 * 32) * sizeof(__bf16);
  (void)hipFuncSetAttribute((const void*)wa3d_attn_kernel,
                            hipFuncAttributeMaxDynamicSharedMemorySize, (int)smemA);

  wa3d_attn_kernel<<<dim3(512 * NHEADS), dim3(128), smemA, stream>>>(
      x, mask, qkv_w, qkv_b, bias_table, rel_index, cxt);
  wa3d_proj_kernel<<<dim3(512), dim3(128), 0, stream>>>(
      cxt, proj_w, proj_b, (float*)d_out);
}
